// BinaryLinear_28509992911461
// MI455X (gfx1250) — matrix operand order repaired and checked by reading, not compiled, not run on hardware
//
#include <hip/hip_runtime.h>
#include <hip/hip_bf16.h>

// ---------------------------------------------------------------------------
// BinaryLinear: out = mean(|W|) * (x @ sign(W)^T)
//   x: [16384, 1024] f32, W: [1024, 1024] f32, out: [16384, 1024] f32
//
// Pipeline:
//   1) two-stage reduction -> scale = mean(|W|)
//   2) W  -> bf16 {-1,0,+1}  (2 MB in ws)
//   3) x  -> bf16            (32 MB in ws)   [kills all cvt VALU in GEMM loop]
//   4) WMMA GEMM (v_wmma_f32_16x16x32_bf16), 4x4 tiles/wave, scale in epilogue
// ---------------------------------------------------------------------------

typedef __attribute__((ext_vector_type(16))) __bf16 v16bf;
typedef __attribute__((ext_vector_type(8)))  __bf16 v8bf;
typedef __attribute__((ext_vector_type(8)))  float  v8f;
typedef __attribute__((ext_vector_type(4)))  float  v4f;

#define K_DIM 1024
#define N_DIM 1024

// ---------------- Stage 1: partial sums of |W| (1024 blocks x 1024 elems) ----
__global__ __launch_bounds__(256) void wabs_partial_kernel(
    const float* __restrict__ w, float* __restrict__ partial) {
  __shared__ float sm[256];
  const int tid = threadIdx.x;
  const int base = blockIdx.x * 1024;
  float s = 0.0f;
#pragma unroll
  for (int i = 0; i < 4; ++i) s += fabsf(w[base + tid + i * 256]);
  sm[tid] = s;
  __syncthreads();
  for (int off = 128; off > 0; off >>= 1) {
    if (tid < off) sm[tid] += sm[tid + off];
    __syncthreads();
  }
  if (tid == 0) partial[blockIdx.x] = sm[0];
}

// ---------------- Stage 2: fold 1024 partials -> scale = mean(|W|) ----------
__global__ __launch_bounds__(256) void wabs_final_kernel(
    const float* __restrict__ partial, float* __restrict__ scale_out) {
  __shared__ float sm[256];
  const int tid = threadIdx.x;
  float s = partial[tid] + partial[tid + 256] + partial[tid + 512] + partial[tid + 768];
  sm[tid] = s;
  __syncthreads();
  for (int off = 128; off > 0; off >>= 1) {
    if (tid < off) sm[tid] += sm[tid + off];
    __syncthreads();
  }
  if (tid == 0) scale_out[0] = sm[0] * (1.0f / (1024.0f * 1024.0f));
}

// ---------------- Binarize W -> bf16 {-1, 0, +1} table ----------------------
__global__ __launch_bounds__(256) void binarize_kernel(
    const float* __restrict__ w, __bf16* __restrict__ bw) {
  const int i = blockIdx.x * 256 + threadIdx.x;
  const float v = w[i];
  const float s = (v > 0.0f) ? 1.0f : ((v < 0.0f) ? -1.0f : 0.0f);
  bw[i] = (__bf16)s;
}

// ---------------- Convert x -> bf16 (8 elements per thread) -----------------
__global__ __launch_bounds__(256) void cvt_x_kernel(
    const float* __restrict__ x, __bf16* __restrict__ xb) {
  const size_t i = ((size_t)blockIdx.x * 256 + threadIdx.x) * 8;
  const v4f a = *(const v4f*)(x + i);
  const v4f b = *(const v4f*)(x + i + 4);
  v8bf r;
#pragma unroll
  for (int j = 0; j < 4; ++j) {
    r[j]     = (__bf16)a[j];
    r[4 + j] = (__bf16)b[j];
  }
  *(v8bf*)(xb + i) = r;
}

// ---------------- WMMA GEMM: out[m,n] = scale * sum_k xb[m,k]*sign(W)[n,k] --
// Block: 256 threads = 8 wave32 waves; block tile = 128(M) x 256(N).
// Wave (wm in 0..1, wn in 0..3) owns a 64x64 sub-tile = 4x4 WMMA 16x16 tiles:
// 16 b128 loads per 16 WMMAs per K-step (WMMA:load ratio 1.0).
__global__ __launch_bounds__(256) void bin_gemm_wmma_kernel(
    const __bf16* __restrict__ xb, const __bf16* __restrict__ bw,
    const float* __restrict__ scale_p, float* __restrict__ out) {
  const int lane = threadIdx.x & 31;
  const int wave = threadIdx.x >> 5;
  const int hi   = lane >> 4;     // half-wave select
  const int l15  = lane & 15;     // row/col within tile
  const int wm   = wave & 1;      // M sub-position of this wave
  const int wn   = wave >> 1;     // N sub-position of this wave

  const int mBase = blockIdx.y * 128 + wm * 64;
  const int nBase = blockIdx.x * 256 + wn * 64;

  v8f acc[4][4];
#pragma unroll
  for (int mi = 0; mi < 4; ++mi)
#pragma unroll
    for (int ni = 0; ni < 4; ++ni)
#pragma unroll
      for (int v = 0; v < 8; ++v) acc[mi][ni][v] = 0.0f;

  // A fragment (16-bit 16x32 layout): lanes 0-15 = rows M=0..15; per lane the
  // 16 bf16 elements are two 8-wide K runs at K = hi*8 and K = 16 + hi*8.
  const __bf16* xrow[4];
#pragma unroll
  for (int mi = 0; mi < 4; ++mi)
    xrow[mi] = xb + (size_t)(mBase + mi * 16 + l15) * K_DIM;

  // B fragment (32x16): lane = column N (= row n of W since B = sign(W)^T);
  // per lane two 8-wide K runs at K = hi*8 and K = 16 + hi*8.
  const __bf16* brow[4];
#pragma unroll
  for (int ni = 0; ni < 4; ++ni)
    brow[ni] = bw + (size_t)(nBase + ni * 16 + l15) * K_DIM;

  for (int kb = 0; kb < K_DIM; kb += 32) {
    v16bf a[4];
#pragma unroll
    for (int mi = 0; mi < 4; ++mi) {
      const v8bf lo = *(const v8bf*)(xrow[mi] + kb + hi * 8);
      const v8bf hv = *(const v8bf*)(xrow[mi] + kb + 16 + hi * 8);
      a[mi] = __builtin_shufflevector(lo, hv, 0, 1, 2, 3, 4, 5, 6, 7,
                                      8, 9, 10, 11, 12, 13, 14, 15);
    }

    v16bf b[4];
#pragma unroll
    for (int ni = 0; ni < 4; ++ni) {
      const v8bf blo = *(const v8bf*)(brow[ni] + kb + hi * 8);
      const v8bf bhv = *(const v8bf*)(brow[ni] + kb + 16 + hi * 8);
      b[ni] = __builtin_shufflevector(blo, bhv, 0, 1, 2, 3, 4, 5, 6, 7,
                                      8, 9, 10, 11, 12, 13, 14, 15);
    }

    // Prefetch next K slab of x rows into cache (global_prefetch_b8).
    if (kb + 32 < K_DIM) {
      __builtin_prefetch(xrow[0] + kb + 32 + hi * 8, 0, 1);
      __builtin_prefetch(xrow[2] + kb + 32 + hi * 8, 0, 1);
    }

#pragma unroll
    for (int mi = 0; mi < 4; ++mi)
#pragma unroll
      for (int ni = 0; ni < 4; ++ni)
        acc[mi][ni] = __builtin_amdgcn_wmma_f32_16x16x32_bf16(
            /*neg_a=*/false, a[mi], /*neg_b=*/false, b[ni],
            /*c_mod=*/(short)0, acc[mi][ni],
            /*reuse_a=*/false, /*reuse_b=*/false);
  }

  const float scale = scale_p[0];

  // C/D layout: VGPR v holds M = v + 8*hi, N = l15 within each 16x16 tile.
#pragma unroll
  for (int mi = 0; mi < 4; ++mi) {
#pragma unroll
    for (int ni = 0; ni < 4; ++ni) {
      const int m0 = mBase + mi * 16 + 8 * hi;
      const int n0 = nBase + ni * 16 + l15;
      float* o = out + (size_t)m0 * N_DIM + n0;
#pragma unroll
      for (int v = 0; v < 8; ++v) o[(size_t)v * N_DIM] = acc[mi][ni][v] * scale;
    }
  }
}

// ---------------------------------------------------------------------------
extern "C" void kernel_launch(void* const* d_in, const int* in_sizes, int n_in,
                              void* d_out, int out_size, void* d_ws, size_t ws_size,
                              hipStream_t stream) {
  const float* x = (const float*)d_in[0];        // [M, 1024]
  const float* w = (const float*)d_in[1];        // [1024, 1024]
  float* out = (float*)d_out;                    // [M, 1024]

  const int M = in_sizes[0] / K_DIM;             // 16384

  // Workspace layout:
  //   [0, 32MB)           : x converted to bf16          (M*1024)
  //   [32MB, 34MB)        : binarized W as bf16          (1024*1024)
  //   [34MB, 34MB+4KB)    : 1024 partial |W| sums (f32)
  //   [34MB+4KB, ...)     : scale scalar (f32)
  char* ws = (char*)d_ws;
  const size_t xb_bytes = (size_t)M * K_DIM * sizeof(__bf16);
  __bf16* xbuf    = (__bf16*)ws;
  __bf16* bw      = (__bf16*)(ws + xb_bytes);
  float* partials = (float*)(ws + xb_bytes + (size_t)2 * 1024 * 1024);
  float* scale_p  = (float*)(ws + xb_bytes + (size_t)2 * 1024 * 1024 + 4096);

  wabs_partial_kernel<<<1024, 256, 0, stream>>>(w, partials);
  wabs_final_kernel<<<1, 256, 0, stream>>>(partials, scale_p);
  binarize_kernel<<<(1024 * 1024) / 256, 256, 0, stream>>>(w, bw);

  const size_t xElems = (size_t)M * K_DIM;
  cvt_x_kernel<<<(unsigned)(xElems / (256 * 8)), 256, 0, stream>>>(x, xbuf);

  dim3 grid(N_DIM / 256, M / 128);   // (4, 128)
  bin_gemm_wmma_kernel<<<grid, 256, 0, stream>>>(xbuf, bw, scale_p, out);
}
